// Model_38912403702173
// MI455X (gfx1250) — matrix operand order repaired and checked by reading, not compiled, not run on hardware
//
#include <hip/hip_runtime.h>
#include <hip/hip_bf16.h>

typedef __attribute__((ext_vector_type(16))) __bf16 v16bf;
typedef __attribute__((ext_vector_type(8)))  float  v8f;

#define BB     512
#define LL     100
#define DD     128
#define NITEMS 100000
#define NODES  112            // 100 padded to 7*16
#define NEGV   (-1.0e9f)

// ---------------------------------------------------------------------------
// Workspace layout (bytes, all 256-aligned)
// ---------------------------------------------------------------------------
#define O_EMBBF   0u                       // 100000*128*2 = 25,600,000
#define O_WIN     25600000u                // 128*128*2    = 32,768
#define O_WOUT    25632768u
#define O_WIH     25665536u                // 384*128*2    = 98,304
#define O_WHH     25763840u
#define O_HREAD   25862144u                // 512*128*2    = 131,072
#define O_FLAG    25993216u                // 512*4
#define O_NODES   25995264u                // 512*112*4    = 229,376
#define O_META    26224640u                // 512*4*4
#define O_CINV    26232832u                // 512*100*4    = 204,800

// ---------------------------------------------------------------------------
// f32 -> bf16 conversion
// ---------------------------------------------------------------------------
__global__ void k_cvt(const float* __restrict__ src, __bf16* __restrict__ dst, int n) {
    int i = blockIdx.x * 256 + threadIdx.x;
    if (i < n) dst[i] = (__bf16)src[i];
}

// ---------------------------------------------------------------------------
// Per-sample graph preprocessing: unique nodes, token->node map, counts
// ---------------------------------------------------------------------------
__global__ __launch_bounds__(128) void k_prep(const int* __restrict__ x,
                                              int* __restrict__ nodes,
                                              int* __restrict__ meta,
                                              int* __restrict__ cinv) {
    int b = blockIdx.x, t = threadIdx.x;
    __shared__ int sSeq[LL], sFirst[LL], sNode[LL];
    __shared__ int sN, sCnt;
    if (t < LL) sSeq[t] = x[b * LL + t];
    __syncthreads();
    if (t < LL) {
        int v = sSeq[t];
        int first = (v != 0);
        for (int i = 0; i < t; ++i) if (sSeq[i] == v) { first = 0; break; }
        sFirst[t] = first;
    }
    __syncthreads();
    if (t < LL) {
        int v = sSeq[t], r = 0;
        if (v != 0) for (int i = 0; i < LL; ++i) r += (sFirst[i] && sSeq[i] < v);
        sNode[t] = (v != 0) ? r : -1;
    }
    __syncthreads();
    if (t == 0) {
        int n = 0, c = 0;
        for (int i = 0; i < LL; ++i) { n += sFirst[i]; c += (sSeq[i] != 0); }
        sN = n; sCnt = c;
    }
    if (t < NODES) nodes[b * NODES + t] = 0;         // pad -> emb row 0 (zeros)
    __syncthreads();
    if (t < LL && sFirst[t]) nodes[b * NODES + sNode[t]] = sSeq[t];
    if (t < LL && sSeq[t] != 0) {
        int p = 0;
        for (int i = 0; i < t; ++i) p += (sSeq[i] != 0);
        cinv[b * LL + p] = sNode[t];
    }
    if (t == 0) {
        int li = 0;
        for (int i = 0; i < LL; ++i) if (sSeq[i] != 0) li = sNode[i];
        meta[b * 4 + 0] = sN;
        meta[b * 4 + 1] = sCnt;
        meta[b * 4 + 2] = li;
        meta[b * 4 + 3] = 0;
    }
}

// ---------------------------------------------------------------------------
// WMMA fragment helpers (bf16 16x16x32, f32 accumulate)
// A: 16x32, lane holds row lane&15; 16 contiguous K, hi-half lanes +16
// B: 32x16, lane holds col lane&15; 16 contiguous K, hi-half lanes +16
// C: element v -> row v + 8*(lane>>4), col lane&15
// ---------------------------------------------------------------------------
__device__ __forceinline__ v16bf load_a_frag(const __bf16* base, int row, int kofs) {
    const __bf16* p = base + row * DD + kofs;
    v16bf a;
#pragma unroll
    for (int h = 0; h < 8; ++h) { a[h] = p[h]; a[h + 8] = p[8 + h]; }
    return a;
}
__device__ __forceinline__ v16bf load_b_frag(const __bf16* base, int nrow, int kofs) {
    const __bf16* p = base + (long)nrow * DD + kofs;
    v16bf bv;
#pragma unroll
    for (int h = 0; h < 16; ++h) bv[h] = p[h];
    return bv;
}
// acc += A(LDS, row-major, stride D) x W^T  (W: N x K row-major bf16 in global)
__device__ __forceinline__ v8f tile_gemm(const __bf16* A, const __bf16* W,
                                         int mt, int nt, int lane, v8f acc) {
    int arow = mt * 16 + (lane & 15);
    int brow = nt * 16 + (lane & 15);
    int ka = (lane >> 4) << 4;
    int kb = (lane >> 4) << 4;
#pragma unroll
    for (int ks = 0; ks < DD; ks += 32) {
        v16bf a  = load_a_frag(A, arow, ks + ka);
        v16bf bv = load_b_frag(W, brow, ks + kb);
        acc = __builtin_amdgcn_wmma_f32_16x16x32_bf16(false, a, false, bv,
                                                      (short)0, acc, false, false);
    }
    return acc;
}

// ---------------------------------------------------------------------------
// Per-sample session kernel: GNN agg + GRU gates + attention readout
// ---------------------------------------------------------------------------
__global__ __launch_bounds__(128) void k_session(
        const __bf16* __restrict__ emb_bf,
        const __bf16* __restrict__ win_bf,  const __bf16* __restrict__ wout_bf,
        const __bf16* __restrict__ wih_bf,  const __bf16* __restrict__ whh_bf,
        const float* __restrict__ b_ih,     const float* __restrict__ b_hh,
        const float* __restrict__ W_read,   const float* __restrict__ b_read,
        const int* __restrict__ nodes, const int* __restrict__ meta,
        const int* __restrict__ cinv,
        __bf16* __restrict__ hread_bf, int* __restrict__ flags) {
    int b = blockIdx.x, tid = threadIdx.x;
    int wave = tid >> 5, lane = tid & 31;

    __shared__ __bf16 sH0[NODES * DD];    // 28,672 B
    __shared__ __bf16 sB1[NODES * DD];    // Hout, then r-gate
    __shared__ __bf16 sXin[NODES * DD];
    __shared__ float  sAgg[NODES * DD];   // 57,344 B; reused as z(bf16)+H(bf16)
    __shared__ float  sDeg[NODES];
    __shared__ int    sCinv[LL];
    __shared__ int    sNod[NODES];
    __shared__ int    sMeta[4];
    __shared__ float  sScore[NODES];
    __shared__ float  sHL[DD], sLoc[DD];
    __shared__ float  sMax, sSum;

    __bf16* sZ = (__bf16*)sAgg;
    __bf16* sH = ((__bf16*)sAgg) + NODES * DD;

    if (tid < 4)  sMeta[tid] = meta[b * 4 + tid];
    if (tid < LL) sCinv[tid] = cinv[b * LL + tid];
    if (tid < NODES) { sNod[tid] = nodes[b * NODES + tid]; sDeg[tid] = 0.f; }
    for (int i = tid; i < NODES * DD; i += 128) sAgg[i] = 0.f;
    __syncthreads();

    // ---------------------------------------------------------------------
    // Gather H0 = emb[nodes] with async Global->LDS copies (ASYNCcnt path).
    // 128 lanes x 16B = 2KB per issue; 8 rows (of 256B) per iteration.
    // Pad rows hit emb row 0 == zeros.
    // ---------------------------------------------------------------------
    {
        int sub = tid & 15;                 // 16B chunk within a 256B row
        int rof = tid >> 4;                 // 8 rows per iteration
#pragma unroll 1
        for (int it = 0; it < NODES / 8; ++it) {
            int row = it * 8 + rof;
            long node = (long)sNod[row];
            const char* gsrc = (const char*)(emb_bf + node * DD) + sub * 16;
            unsigned ldst = (unsigned)(uintptr_t)(&sH0[row * DD]) + sub * 16;
            asm volatile("global_load_async_to_lds_b128 %0, %1, off"
                         :: "v"(ldst), "v"(gsrc) : "memory");
        }
        asm volatile("s_wait_asynccnt 0x0" ::: "memory");
    }
    __syncthreads();

    int nN = sMeta[0], cnt = sMeta[1], li = sMeta[2];

    // Pass 1: Hout = H0 @ W_out^T
    for (int t = wave; t < 56; t += 4) {
        int mt = t >> 3, nt = t & 7;
        v8f acc = {};
        acc = tile_gemm(sH0, wout_bf, mt, nt, lane, acc);
        int col = nt * 16 + (lane & 15);
#pragma unroll
        for (int v = 0; v < 8; ++v) {
            int row = mt * 16 + v + ((lane >> 4) << 3);
            sB1[row * DD + col] = (__bf16)acc[v];
        }
    }
    __syncthreads();

    // Edge scatter: agg[dst] += Hout[src]; deg[dst]++   (column-exclusive per thread)
    int ne = cnt - 1;
    for (int k = 0; k < ne; ++k) {
        int s = sCinv[k], d = sCinv[k + 1];
        sAgg[d * DD + tid] += (float)sB1[s * DD + tid];
        if (tid == 0) sDeg[d] += 1.f;
    }
    __syncthreads();

    // Pass 2: xin = H0 @ W_in^T + agg/deg
    for (int t = wave; t < 56; t += 4) {
        int mt = t >> 3, nt = t & 7;
        v8f acc = {};
        acc = tile_gemm(sH0, win_bf, mt, nt, lane, acc);
        int col = nt * 16 + (lane & 15);
#pragma unroll
        for (int v = 0; v < 8; ++v) {
            int row = mt * 16 + v + ((lane >> 4) << 3);
            float dg = sDeg[row]; dg = dg < 1.f ? 1.f : dg;
            sXin[row * DD + col] = (__bf16)(acc[v] + sAgg[row * DD + col] / dg);
        }
    }
    __syncthreads();

    // Pass 3/4: r and z gates = sigmoid(xin@w_ih_g^T + H0@w_hh_g^T + biases)
    for (int g = 0; g < 2; ++g) {
        for (int t = wave; t < 56; t += 4) {
            int mt = t >> 3, nt = t & 7;
            v8f acc = {};
            acc = tile_gemm(sXin, wih_bf + g * DD * DD, mt, nt, lane, acc);
            acc = tile_gemm(sH0,  whh_bf + g * DD * DD, mt, nt, lane, acc);
            int col  = nt * 16 + (lane & 15);
            float bias = b_ih[g * DD + col] + b_hh[g * DD + col];
            __bf16* dst = (g == 0) ? sB1 : sZ;
#pragma unroll
            for (int v = 0; v < 8; ++v) {
                int row = mt * 16 + v + ((lane >> 4) << 3);
                float s = acc[v] + bias;
                dst[row * DD + col] = (__bf16)(1.f / (1.f + __expf(-s)));
            }
        }
        __syncthreads();
    }

    // Pass 5: nct = tanh(gi_n + r*gh_n); H = (1-z)*nct + z*H0 (masked)
    for (int t = wave; t < 56; t += 4) {
        int mt = t >> 3, nt = t & 7;
        v8f gi = {}, gh = {};
        gi = tile_gemm(sXin, wih_bf + 2 * DD * DD, mt, nt, lane, gi);
        gh = tile_gemm(sH0,  whh_bf + 2 * DD * DD, mt, nt, lane, gh);
        int col = nt * 16 + (lane & 15);
        float bi = b_ih[2 * DD + col], bh = b_hh[2 * DD + col];
#pragma unroll
        for (int v = 0; v < 8; ++v) {
            int row = mt * 16 + v + ((lane >> 4) << 3);
            float r  = (float)sB1[row * DD + col];
            float z  = (float)sZ[row * DD + col];
            float h0 = (float)sH0[row * DD + col];
            float nct = tanhf(gi[v] + bi + r * (gh[v] + bh));
            float h = (1.f - z) * nct + z * h0;
            if (row >= nN) h = 0.f;
            sH[row * DD + col] = (__bf16)h;
        }
    }
    __syncthreads();

    // Attention readout
    if (tid < DD) sHL[tid] = (float)sH[li * DD + tid];
    __syncthreads();
    if (tid < NODES) {
        float s = NEGV;
        if (tid < nN) {
            s = 0.f;
            for (int d = 0; d < DD; ++d) s += (float)sH[tid * DD + d] * sHL[d];
        }
        sScore[tid] = s;
    }
    __syncthreads();
    if (tid == 0) {
        float m = -1.0e30f;
        for (int i = 0; i < NODES; ++i) m = fmaxf(m, sScore[i]);
        sMax = m;
    }
    __syncthreads();
    if (tid < NODES) sScore[tid] = __expf(sScore[tid] - sMax);
    __syncthreads();
    if (tid == 0) {
        float s = 0.f;
        for (int i = 0; i < NODES; ++i) s += sScore[i];
        sSum = s;
    }
    __syncthreads();
    if (tid < DD) {
        float acc = 0.f;
        for (int i = 0; i < nN; ++i) acc += sScore[i] * (float)sH[i * DD + tid];
        sLoc[tid] = acc / sSum;
    }
    __syncthreads();
    if (tid < DD) {
        float a = b_read[tid];
        const float* wr = W_read + tid * (3 * DD);
        for (int j = 0; j < DD; ++j) a += wr[j] * sHL[j] + wr[DD + j] * sLoc[j];
        hread_bf[b * DD + tid] = (__bf16)tanhf(a);
    }
    if (tid == 0) flags[b] = (cnt >= 2) ? 1 : 0;
}

// ---------------------------------------------------------------------------
// Final scoring: out[b][i] = flag ? h_read[b] . emb[i] : (i==0 ? 0 : NEG)
// Each wave owns a 16-item tile; emb B-frags loaded once, loop 32 M-tiles.
// ---------------------------------------------------------------------------
__global__ __launch_bounds__(128) void k_score(const __bf16* __restrict__ hread,
                                               const __bf16* __restrict__ embbf,
                                               const int* __restrict__ flags,
                                               float* __restrict__ out) {
    __shared__ int sFlag[BB];
    int tid = threadIdx.x;
    for (int i = tid; i < BB; i += 128) sFlag[i] = flags[i];
    __syncthreads();

    int wave = tid >> 5, lane = tid & 31;
    long it0 = (long)blockIdx.x * 64 + wave * 16;
    if (it0 >= NITEMS) return;                       // wave-uniform

    int ncol = (int)it0 + (lane & 15);
    v16bf bfr[4];
#pragma unroll
    for (int ks = 0; ks < 4; ++ks)
        bfr[ks] = load_b_frag(embbf, ncol, ks * 32 + ((lane >> 4) << 4));

    for (int mt = 0; mt < 32; ++mt) {
        v8f acc = {};
#pragma unroll
        for (int ks = 0; ks < 4; ++ks) {
            v16bf a = load_a_frag(hread, mt * 16 + (lane & 15),
                                  ks * 32 + ((lane >> 4) << 4));
            acc = __builtin_amdgcn_wmma_f32_16x16x32_bf16(false, a, false, bfr[ks],
                                                          (short)0, acc, false, false);
        }
#pragma unroll
        for (int v = 0; v < 8; ++v) {
            int row = mt * 16 + v + ((lane >> 4) << 3);
            float val = sFlag[row] ? acc[v] : ((ncol == 0) ? 0.f : NEGV);
            out[(long)row * NITEMS + ncol] = val;
        }
    }
}

// ---------------------------------------------------------------------------
extern "C" void kernel_launch(void* const* d_in, const int* in_sizes, int n_in,
                              void* d_out, int out_size, void* d_ws, size_t ws_size,
                              hipStream_t stream) {
    const int*   x      = (const int*)d_in[0];
    const float* emb    = (const float*)d_in[2];
    const float* W_in   = (const float*)d_in[3];
    const float* W_out  = (const float*)d_in[4];
    const float* w_ih   = (const float*)d_in[5];
    const float* w_hh   = (const float*)d_in[6];
    const float* b_ih   = (const float*)d_in[7];
    const float* b_hh   = (const float*)d_in[8];
    const float* W_read = (const float*)d_in[9];
    const float* b_read = (const float*)d_in[10];

    char* ws = (char*)d_ws;
    __bf16* emb_bf   = (__bf16*)(ws + O_EMBBF);
    __bf16* win_bf   = (__bf16*)(ws + O_WIN);
    __bf16* wout_bf  = (__bf16*)(ws + O_WOUT);
    __bf16* wih_bf   = (__bf16*)(ws + O_WIH);
    __bf16* whh_bf   = (__bf16*)(ws + O_WHH);
    __bf16* hread_bf = (__bf16*)(ws + O_HREAD);
    int*    flagsW   = (int*)   (ws + O_FLAG);
    int*    nodesW   = (int*)   (ws + O_NODES);
    int*    metaW    = (int*)   (ws + O_META);
    int*    cinvW    = (int*)   (ws + O_CINV);

    k_cvt<<<(NITEMS * DD + 255) / 256, 256, 0, stream>>>(emb,   emb_bf,  NITEMS * DD);
    k_cvt<<<(DD * DD + 255) / 256,     256, 0, stream>>>(W_in,  win_bf,  DD * DD);
    k_cvt<<<(DD * DD + 255) / 256,     256, 0, stream>>>(W_out, wout_bf, DD * DD);
    k_cvt<<<(3 * DD * DD + 255) / 256, 256, 0, stream>>>(w_ih,  wih_bf,  3 * DD * DD);
    k_cvt<<<(3 * DD * DD + 255) / 256, 256, 0, stream>>>(w_hh,  whh_bf,  3 * DD * DD);

    k_prep<<<BB, 128, 0, stream>>>(x, nodesW, metaW, cinvW);

    k_session<<<BB, 128, 0, stream>>>(emb_bf, win_bf, wout_bf, wih_bf, whh_bf,
                                      b_ih, b_hh, W_read, b_read,
                                      nodesW, metaW, cinvW, hread_bf, flagsW);

    k_score<<<(NITEMS + 63) / 64, 128, 0, stream>>>(hread_bf, emb_bf, flagsW,
                                                    (float*)d_out);
}
